// AdaptiveThresholding_55224689492343
// MI455X (gfx1250) — matrix operand order repaired and checked by reading, not compiled, not run on hardware
//
#include <hip/hip_runtime.h>

typedef __attribute__((ext_vector_type(8)))  _Float16 v8h;
typedef __attribute__((ext_vector_type(16))) _Float16 v16h;
typedef __attribute__((ext_vector_type(8)))  float    v8f;

#define H_IMG   512
#define W_IMG   512
#define TILE_W  64      // output cols per workgroup
#define TILE_H  32      // output rows per workgroup
#define IN_W    80      // halo tile cols  (64 + 2*8)
#define IN_H    48      // halo tile rows  (32 + 2*8)
#define IN_PITCH 88     // halves; multiple of 8 -> 16B-aligned b128 rows, conflict-free
#define HS_PITCH 40     // halves per column of per-wave H scratch; multiple of 8
#define NWAVES   8

static __device__ __forceinline__ v16h cat16(v8h lo, v8h hi) {
    v16h r;
#pragma unroll
    for (int i = 0; i < 8; ++i) { r[i] = lo[i]; r[i + 8] = hi[i]; }
    return r;
}

__global__ __launch_bounds__(256)
void AdaptiveThreshold_wmma(const float* __restrict__ in,
                            float* __restrict__ out)
{
    __shared__ __attribute__((aligned(16))) _Float16 sIn[IN_H * IN_PITCH];          // 8448 B
    __shared__ __attribute__((aligned(16))) _Float16 sH[NWAVES * 16 * HS_PITCH];    // 10240 B

    const int tid = threadIdx.x;
    const int z   = blockIdx.z;
    const int gr0 = blockIdx.y * TILE_H;
    const int gc0 = blockIdx.x * TILE_W;
    const float* img = in + (size_t)z * (H_IMG * W_IMG);

    // ---- Stage 0: cooperative halo load, f32 -> f16, replicate-clamped ----
    for (int e = tid; e < IN_H * IN_W; e += 256) {
        int r = e / IN_W;
        int c = e - r * IN_W;
        int gr = gr0 - 8 + r; gr = gr < 0 ? 0 : (gr > H_IMG - 1 ? H_IMG - 1 : gr);
        int gc = gc0 - 8 + c; gc = gc < 0 ? 0 : (gc > W_IMG - 1 ? W_IMG - 1 : gc);
        sIn[r * IN_PITCH + c] = (_Float16)img[gr * W_IMG + gc];
    }
    __syncthreads();

    // ---- Per-wave 16x16 output tile ----
    const int lane = tid & 31;
    const int w    = tid >> 5;
    const int tc   = w & 3;       // 4 tiles across -> 64 cols
    const int tr   = w >> 2;      // 2 tiles down   -> 32 rows
    const int n    = lane & 15;   // frag col (B/C/D) or frag row (A)
    const int hi   = lane >> 4;

    // Horizontal band, B-fragment layout (32x16): half i <-> K = (i<8?0:16)+hi*8+(i&7), col n
    v16h bh;
#pragma unroll
    for (int i = 0; i < 16; ++i) {
        int k = (i < 8 ? 0 : 16) + hi * 8 + (i & 7);
        bh[i] = (k >= n + 3 && k <= n + 13) ? (_Float16)1.0f : (_Float16)0.0f;
    }
    // Vertical band, A-fragment layout (16x32): half i <-> K = hi*16 + i, row n
    v16h bv;
#pragma unroll
    for (int i = 0; i < 16; ++i) {
        int k = hi * 16 + i;
        bv[i] = (k >= n + 3 && k <= n + 13) ? (_Float16)1.0f : (_Float16)0.0f;
    }

    // ---- Stage 1: two horizontal WMMAs (rows tr*16 .. tr*16+31 of halo tile) ----
    v8f h0, h1;
    {
        const int colb = tc * 16 + hi * 8;   // halves; *2 bytes -> 16B aligned
#pragma unroll
        for (int b = 0; b < 2; ++b) {
            const int row = tr * 16 + b * 16 + n;
            v8h lo = *(const v8h*)&sIn[row * IN_PITCH + colb];
            v8h hi8 = *(const v8h*)&sIn[row * IN_PITCH + colb + 16];
            v16h a = cat16(lo, hi8);
            v8f acc = {};
            v8f d = __builtin_amdgcn_wmma_f32_16x16x32_f16(
                        false, a, false, bh, (short)0, acc, false, false);
            if (b == 0) h0 = d; else h1 = d;
        }
    }

    // ---- Transpose H (C-layout f32) -> B-layout f16 via per-wave LDS scratch ----
    {
        const int base = w * (16 * HS_PITCH) + n * HS_PITCH;  // column n, transposed
        v8h p0, p1;
#pragma unroll
        for (int i = 0; i < 8; ++i) { p0[i] = (_Float16)h0[i]; p1[i] = (_Float16)h1[i]; }
        *(v8h*)&sH[base + 0 * 16 + hi * 8] = p0;   // k = v + 8*hi       (block 0)
        *(v8h*)&sH[base + 1 * 16 + hi * 8] = p1;   // k = 16 + v + 8*hi  (block 1)
    }
    asm volatile("s_wait_dscnt 0" ::: "memory");   // same-wave DS in-order; fence compiler too
    v16h hfrag;
    {
        const int base = w * (16 * HS_PITCH) + n * HS_PITCH + hi * 16;
        v8h lo = *(const v8h*)&sH[base];
        v8h hi8 = *(const v8h*)&sH[base + 8];
        hfrag = cat16(lo, hi8);
    }

    // ---- Stage 2: vertical WMMA -> 11x11 box sums ----
    v8f acc = {};
    v8f s = __builtin_amdgcn_wmma_f32_16x16x32_f16(
                false, bv, false, hfrag, (short)0, acc, false, false);

    // ---- Compare against exact f32 input (cache-hot) and store ----
    const int orow0 = gr0 + tr * 16;
    const int ocol  = gc0 + tc * 16 + n;
#pragma unroll
    for (int v = 0; v < 8; ++v) {
        int row = orow0 + v + 8 * hi;                       // C-layout: M = v + 8*(lane>>4)
        size_t idx = (size_t)z * (H_IMG * W_IMG) + (size_t)row * W_IMG + ocol;
        float x = in[idx];
        float t = s[v] * (1.0f / 121.0f) - 0.02f;           // local mean - CONSTANT
        out[idx] = (x > t) ? 0.0f : 1.0f;
    }
}

extern "C" void kernel_launch(void* const* d_in, const int* in_sizes, int n_in,
                              void* d_out, int out_size, void* d_ws, size_t ws_size,
                              hipStream_t stream) {
    (void)n_in; (void)out_size; (void)d_ws; (void)ws_size;
    const float* in = (const float*)d_in[0];
    float* out = (float*)d_out;
    const int batch = in_sizes[0] / (H_IMG * W_IMG);
    dim3 grid(W_IMG / TILE_W, H_IMG / TILE_H, batch);
    AdaptiveThreshold_wmma<<<grid, dim3(256), 0, stream>>>(in, out);
}
